// MSAColumnGlobalAttention_52802327937520
// MI455X (gfx1250) — matrix operand order repaired and checked by reading, not compiled, not run on hardware
//
#include <hip/hip_runtime.h>
#include <hip/hip_bf16.h>

// Problem constants
#define SS 512   // n_seqs
#define II 512   // n_res
#define CC 256   // c_in
#define HH 8
#define DD 32
#define TDm 256  // H*D

typedef __attribute__((ext_vector_type(16))) __bf16 v16bf;
typedef __attribute__((ext_vector_type(8)))  float  v8f;
typedef __attribute__((ext_vector_type(4)))  float  f32x4;

union FragU { v16bf bf; f32x4 f4[2]; };

// ---- LDS layout (bytes) ----
#define XLN_STRIDE 264              // bf16 elems per row (16B aligned, bank-staggered)
#define GATED_STRIDE 264
#define SC_STRIDE 9                 // floats per score row (pad to avoid bank conflicts)
#define XLN_OFF   0
#define GATED_OFF (XLN_OFF + SS*XLN_STRIDE*2)        // 270336
#define SC_OFF    (GATED_OFF + 32*GATED_STRIDE*2)    // 287232
#define KQ_OFF    (SC_OFF + SS*SC_STRIDE*4)          // 305664  (kq, reused as av)
#define XBAR_OFF  (KQ_OFF + HH*CC*4)                 // 313856
#define QW_OFF    (XBAR_OFF + CC*4)                  // 314880  (q, reused as weighted)
#define MS_OFF    (QW_OFF + TDm*4)                   // 315904
#define LDS_BYTES (MS_OFF + 16)                      // 315920 <= 320KB

static_assert(LDS_BYTES <= 320*1024, "LDS over budget");

__device__ __forceinline__ unsigned short f2bf(float f) {
  unsigned u = __builtin_bit_cast(unsigned, f);
  u = (u + 0x7fffu + ((u >> 16) & 1u)) >> 16;
  return (unsigned short)u;
}
__device__ __forceinline__ float bfl(unsigned u) { return __builtin_bit_cast(float, u << 16); }
__device__ __forceinline__ float bfh(unsigned u) { return __builtin_bit_cast(float, u & 0xffff0000u); }

__device__ __forceinline__ float wred_sum(float v) {
  #pragma unroll
  for (int m = 16; m > 0; m >>= 1) v += __shfl_xor(v, m, 32);
  return v;
}
__device__ __forceinline__ float wred_max(float v) {
  #pragma unroll
  for (int m = 16; m > 0; m >>= 1) v = fmaxf(v, __shfl_xor(v, m, 32));
  return v;
}

// ---------------------------------------------------------------------------
// Kernel 0: pack Wg and Wo (fp32 row-major [256,256]) into bf16 WMMA B-fragment
// order: [ks 0..7][ntile 0..15][lane 0..31][elem 0..15].
// Lane L holds column n = nt*16 + (L&15); K range = ks*32 + (L<16 ? 0..15 : 16..31).
// ---------------------------------------------------------------------------
__global__ void pack_weights_kernel(const float* __restrict__ Wg,
                                    const float* __restrict__ Wo,
                                    unsigned short* __restrict__ dst) {
  int idx  = blockIdx.x * 256 + threadIdx.x;   // 0 .. 131071
  int e    = idx & 15;
  int lane = (idx >> 4) & 31;
  int nt   = (idx >> 9) & 15;
  int ks   = (idx >> 13) & 7;
  int m    = idx >> 16;                        // 0 = Wg, 1 = Wo
  int k = ks * 32 + ((lane < 16) ? 0 : 16) + e;
  int n = nt * 16 + (lane & 15);
  const float* W = m ? Wo : Wg;
  dst[idx] = f2bf(W[k * 256 + n]);
}

// ---------------------------------------------------------------------------
// Main kernel: one block per column i. 256 threads = 8 wave32.
// ---------------------------------------------------------------------------
__global__ __launch_bounds__(256) void msa_col_attn_kernel(
    const float* __restrict__ x,     const float* __restrict__ mask,
    const float* __restrict__ gamma, const float* __restrict__ beta,
    const float* __restrict__ Wq,    const float* __restrict__ Wk,
    const float* __restrict__ Wv,    const float* __restrict__ bg,
    const float* __restrict__ bo,
    const unsigned short* __restrict__ WgP,
    const unsigned short* __restrict__ WoP,
    float* __restrict__ out)
{
  extern __shared__ char smem[];
  unsigned short* xln   = (unsigned short*)(smem + XLN_OFF);
  unsigned short* gated = (unsigned short*)(smem + GATED_OFF);
  float* sc   = (float*)(smem + SC_OFF);
  float* kq   = (float*)(smem + KQ_OFF);    // later reused as av
  float* xbar = (float*)(smem + XBAR_OFF);
  float* qw   = (float*)(smem + QW_OFF);    // q, later reused as weighted_flat
  float* msum = (float*)(smem + MS_OFF);

  const int i    = blockIdx.x;
  const int tid  = threadIdx.x;
  const int w    = tid >> 5;
  const int lane = tid & 31;

  xbar[tid] = 0.f;
  if (tid == 0) msum[0] = 0.f;
  __syncthreads();

  // ---------------- Phase 1: LayerNorm -> bf16 slab in LDS + masked mean ----
  {
    const int c0 = lane * 8;
    float gg[8], bb[8];
    #pragma unroll
    for (int j = 0; j < 8; j++) { gg[j] = gamma[c0 + j]; bb[j] = beta[c0 + j]; }
    float xacc[8] = {0.f,0.f,0.f,0.f,0.f,0.f,0.f,0.f};
    float mAcc = 0.f;
    for (int r = w; r < SS; r += 8) {
      const float* row = x + ((size_t)(r * II + i)) * CC;
      __builtin_prefetch(row + 8 * (size_t)II * CC + c0, 0, 1); // next row for this wave
      f32x4 a0 = *(const f32x4*)(row + c0);
      f32x4 a1 = *(const f32x4*)(row + c0 + 4);
      float s = 0.f, s2 = 0.f;
      #pragma unroll
      for (int j = 0; j < 4; j++) { s += a0[j] + a1[j]; s2 += a0[j]*a0[j] + a1[j]*a1[j]; }
      s  = wred_sum(s);
      s2 = wred_sum(s2);
      float mu   = s * (1.f / 256.f);
      float var  = s2 * (1.f / 256.f) - mu * mu;
      float rstd = rsqrtf(var + 1e-5f);
      float m = mask[r * II + i];
      float vv[8];
      #pragma unroll
      for (int j = 0; j < 8; j++) {
        float a = (j < 4) ? a0[j] : a1[j - 4];
        float v = (a - mu) * rstd * gg[j] + bb[j];
        vv[j] = v;
        xacc[j] += m * v;
      }
      uint4 st;
      st.x = (unsigned)f2bf(vv[0]) | ((unsigned)f2bf(vv[1]) << 16);
      st.y = (unsigned)f2bf(vv[2]) | ((unsigned)f2bf(vv[3]) << 16);
      st.z = (unsigned)f2bf(vv[4]) | ((unsigned)f2bf(vv[5]) << 16);
      st.w = (unsigned)f2bf(vv[6]) | ((unsigned)f2bf(vv[7]) << 16);
      *(uint4*)(xln + r * XLN_STRIDE + c0) = st;
      if (lane == 0) mAcc += m;
    }
    #pragma unroll
    for (int j = 0; j < 8; j++) atomicAdd(&xbar[c0 + j], xacc[j]);
    if (lane == 0) atomicAdd(msum, mAcc);
  }
  __syncthreads();

  // ---------------- Phase 2: q = (xbar/clip(msum)) @ Wq; kq = (q@Wk^T)/sqrt(D)
  {
    float inv = 1.f / fmaxf(msum[0], 1.f);
    float acc = 0.f;
    for (int c = 0; c < CC; c++) acc += xbar[c] * Wq[c * TDm + tid];
    qw[tid] = acc * inv;
  }
  __syncthreads();
  {
    const int h = tid >> 5;
    const int c0 = lane * 8;
    float qh[DD];
    #pragma unroll
    for (int d = 0; d < DD; d++) qh[d] = qw[h * DD + d];
    const float scale = 0.17677669529663687f; // 1/sqrt(32)
    #pragma unroll
    for (int j = 0; j < 8; j++) {
      int c = c0 + j;
      float s = 0.f;
      #pragma unroll
      for (int d = 0; d < DD; d++) s += qh[d] * Wk[c * DD + d];
      kq[h * CC + c] = s * scale;
    }
  }
  __syncthreads();

  // ---------------- Phase 3: scores[t,h] = xln[t,:]·kq[h,:] + mask bias -----
  {
    const int h  = tid & 7;
    const int tb = tid >> 3;
    for (int it = 0; it < 16; it++) {
      int t = tb + 32 * it;
      const unsigned short* xr = xln + t * XLN_STRIDE;
      const float* kr = kq + h * CC;
      float acc = 0.f;
      for (int c = 0; c < CC; c += 8) {
        uint4 u = *(const uint4*)(xr + c);
        acc += bfl(u.x) * kr[c+0] + bfh(u.x) * kr[c+1]
             + bfl(u.y) * kr[c+2] + bfh(u.y) * kr[c+3]
             + bfl(u.z) * kr[c+4] + bfh(u.z) * kr[c+5]
             + bfl(u.w) * kr[c+6] + bfh(u.w) * kr[c+7];
      }
      float m = mask[t * II + i];
      sc[t * SC_STRIDE + h] = acc + (1.f - m) * (-1e9f);
    }
  }
  __syncthreads();

  // ---------------- Phase 4: softmax over t, one wave per head --------------
  {
    const int h = w;
    float v[16];
    float mx = -3.4e38f;
    #pragma unroll
    for (int k = 0; k < 16; k++) { v[k] = sc[(lane + 32*k) * SC_STRIDE + h]; mx = fmaxf(mx, v[k]); }
    mx = wred_max(mx);
    float sum = 0.f;
    #pragma unroll
    for (int k = 0; k < 16; k++) { v[k] = __expf(v[k] - mx); sum += v[k]; }
    sum = wred_sum(sum);
    float inv = 1.f / sum;
    #pragma unroll
    for (int k = 0; k < 16; k++) sc[(lane + 32*k) * SC_STRIDE + h] = v[k] * inv;
  }
  __syncthreads();

  // ---------------- Phase 5: av = attn^T @ xln; weighted = av @ Wv ----------
  {
    const int h  = tid >> 5;
    const int c0 = lane * 8;
    float acc[8] = {0.f,0.f,0.f,0.f,0.f,0.f,0.f,0.f};
    for (int t = 0; t < SS; t++) {
      float a = sc[t * SC_STRIDE + h];
      uint4 u = *(const uint4*)(xln + t * XLN_STRIDE + c0);
      acc[0] += a * bfl(u.x); acc[1] += a * bfh(u.x);
      acc[2] += a * bfl(u.y); acc[3] += a * bfh(u.y);
      acc[4] += a * bfl(u.z); acc[5] += a * bfh(u.z);
      acc[6] += a * bfl(u.w); acc[7] += a * bfh(u.w);
    }
    #pragma unroll
    for (int j = 0; j < 8; j++) kq[h * CC + c0 + j] = acc[j];  // kq reused as av
  }
  __syncthreads();
  {
    const int h = tid >> 5, d = tid & 31;
    float s = 0.f;
    for (int c = 0; c < CC; c++) s += kq[h * CC + c] * Wv[c * DD + d];
    qw[tid] = s;  // weighted_flat[e], e = h*32+d
  }
  __syncthreads();

  // ---------------- Phase 6: fused gate GEMM + output GEMM (WMMA bf16) ------
  // Per 32-row chunk: G1 = sigmoid(xln@Wg + bg) * weighted -> LDS bf16,
  //                   out = (G1@Wo + bo) * mask.
  // Wave w owns n-tiles {2w, 2w+1} and BOTH 16-row m-tiles of the chunk.
  // Persistent B fragments: 2 nt x 8 ks x 2 GEMMs = 32 frags = 256 VGPRs,
  // deliberately preloaded once so the chunk loop never touches memory for B.
  {
    const int nlo = lane & 15;
    const int hi  = lane >> 4;    // 0/1

    FragU Bg[2][8], Bo[2][8];
    #pragma unroll
    for (int j = 0; j < 2; j++) {
      #pragma unroll
      for (int ks = 0; ks < 8; ks++) {
        int nt = 2 * w + j;
        size_t fo = ((size_t)((ks * 16 + nt) * 32 + lane)) * 2;
        const f32x4* bp = (const f32x4*)WgP + fo;
        Bg[j][ks].f4[0] = bp[0];
        Bg[j][ks].f4[1] = bp[1];
        const f32x4* op = (const f32x4*)WoP + fo;
        Bo[j][ks].f4[0] = op[0];
        Bo[j][ks].f4[1] = op[1];
      }
    }

    float bge[2], wte[2], boe[2];
    #pragma unroll
    for (int j = 0; j < 2; j++) {
      int e = (2 * w + j) * 16 + nlo;
      bge[j] = bg[e];
      wte[j] = qw[e];
      boe[j] = bo[e];
    }

    for (int c0 = 0; c0 < SS; c0 += 32) {
      // ---- GEMM1: xln[c0..c0+31, :] @ Wg ----
      v8f acc[2][2] = {{v8f{}, v8f{}}, {v8f{}, v8f{}}};
      #pragma unroll
      for (int ks = 0; ks < 8; ks++) {
        FragU a0, a1;
        const unsigned short* ap0 = xln + (c0 + nlo) * XLN_STRIDE + ks * 32 + hi * 16;
        const unsigned short* ap1 = ap0 + 16 * XLN_STRIDE;
        a0.f4[0] = *(const f32x4*)ap0;
        a0.f4[1] = *(const f32x4*)(ap0 + 8);
        a1.f4[0] = *(const f32x4*)ap1;
        a1.f4[1] = *(const f32x4*)(ap1 + 8);
        #pragma unroll
        for (int j = 0; j < 2; j++) {
          acc[0][j] = __builtin_amdgcn_wmma_f32_16x16x32_bf16(
              false, a0.bf, false, Bg[j][ks].bf, (short)0, acc[0][j], false, false);
          acc[1][j] = __builtin_amdgcn_wmma_f32_16x16x32_bf16(
              false, a1.bf, false, Bg[j][ks].bf, (short)0, acc[1][j], false, false);
        }
      }
      // epilogue 1: bias + sigmoid + gate*weighted -> gated LDS (bf16)
      #pragma unroll
      for (int mt = 0; mt < 2; mt++) {
        #pragma unroll
        for (int j = 0; j < 2; j++) {
          int e = (2 * w + j) * 16 + nlo;
          #pragma unroll
          for (int r = 0; r < 8; r++) {
            int rl = mt * 16 + r + hi * 8;
            float xg = acc[mt][j][r] + bge[j];
            float g = 1.f / (1.f + __expf(-xg));
            gated[rl * GATED_STRIDE + e] = f2bf(g * wte[j]);
          }
        }
      }
      __syncthreads();

      // ---- GEMM2: gated @ Wo ----
      v8f acc2[2][2] = {{v8f{}, v8f{}}, {v8f{}, v8f{}}};
      #pragma unroll
      for (int ks = 0; ks < 8; ks++) {
        FragU a0, a1;
        const unsigned short* ap0 = gated + nlo * GATED_STRIDE + ks * 32 + hi * 16;
        const unsigned short* ap1 = ap0 + 16 * GATED_STRIDE;
        a0.f4[0] = *(const f32x4*)ap0;
        a0.f4[1] = *(const f32x4*)(ap0 + 8);
        a1.f4[0] = *(const f32x4*)ap1;
        a1.f4[1] = *(const f32x4*)(ap1 + 8);
        #pragma unroll
        for (int j = 0; j < 2; j++) {
          acc2[0][j] = __builtin_amdgcn_wmma_f32_16x16x32_bf16(
              false, a0.bf, false, Bo[j][ks].bf, (short)0, acc2[0][j], false, false);
          acc2[1][j] = __builtin_amdgcn_wmma_f32_16x16x32_bf16(
              false, a1.bf, false, Bo[j][ks].bf, (short)0, acc2[1][j], false, false);
        }
      }
      // epilogue 2: + bo, * mask, store fp32
      #pragma unroll
      for (int mt = 0; mt < 2; mt++) {
        float mrow[8];
        #pragma unroll
        for (int r = 0; r < 8; r++) {
          int s = c0 + mt * 16 + r + hi * 8;
          mrow[r] = mask[s * II + i];
        }
        #pragma unroll
        for (int j = 0; j < 2; j++) {
          int e = (2 * w + j) * 16 + nlo;
          #pragma unroll
          for (int r = 0; r < 8; r++) {
            int s = c0 + mt * 16 + r + hi * 8;
            out[((size_t)(s * II + i)) * CC + e] = (acc2[mt][j][r] + boe[j]) * mrow[r];
          }
        }
      }
      __syncthreads();  // gated buffer reused next chunk
    }
  }
}

extern "C" void kernel_launch(void* const* d_in, const int* in_sizes, int n_in,
                              void* d_out, int out_size, void* d_ws, size_t ws_size,
                              hipStream_t stream) {
  (void)in_sizes; (void)n_in; (void)out_size; (void)ws_size;
  const float* x     = (const float*)d_in[0];
  const float* mask  = (const float*)d_in[1];
  const float* gamma = (const float*)d_in[2];
  const float* beta  = (const float*)d_in[3];
  const float* Wq    = (const float*)d_in[4];
  const float* Wk    = (const float*)d_in[5];
  const float* Wv    = (const float*)d_in[6];
  const float* Wg    = (const float*)d_in[7];
  const float* bg    = (const float*)d_in[8];
  const float* Wo    = (const float*)d_in[9];
  const float* bo    = (const float*)d_in[10];
  unsigned short* WgP = (unsigned short*)d_ws;
  unsigned short* WoP = WgP + 65536;
  float* out = (float*)d_out;

  // Pack Wg/Wo into bf16 WMMA-B fragment layout (L2-resident, 256 KB total).
  pack_weights_kernel<<<512, 256, 0, stream>>>(Wg, Wo, WgP);

  // Main fused kernel: one block per column, ~316 KB dynamic LDS.
  hipFuncSetAttribute((const void*)msa_col_attn_kernel,
                      hipFuncAttributeMaxDynamicSharedMemorySize, (int)LDS_BYTES);
  msa_col_attn_kernel<<<512, 256, LDS_BYTES, stream>>>(
      x, mask, gamma, beta, Wq, Wk, Wv, bg, bo, WgP, WoP, out);
}
